// ScaledTernaryMatmul_90434831385364
// MI455X (gfx1250) — matrix operand order repaired and checked by reading, not compiled, not run on hardware
//
#include <hip/hip_runtime.h>
#include <stdint.h>

// ---------------------------------------------------------------------------
// ScaledTernaryMatmul for MI455X (gfx1250, wave32, WMMA)
//   y[b,u] = clip( (sum_d x[b,d] * tern(pw[u], d)) * scale[u] + bias[u], +-100 )
// One wave owns a 16-unit N-tile and both 16-row M-tiles (B=32): the 58.7 MB
// int32-packed weight stream (the bandwidth limiter at 23.3 TB/s) is read
// exactly once, through a 4-deep register prefetch pipeline. Crumbs decode to
// f16 {0,+1,-1} with a magic-multiply bit spread + v_perm byte LUT into two
// alternating B buffers (kills the WMMA->VALU WAR hazard NOPs); x is staged
// f32->f16 into LDS in a contiguous-K A fragment order (k = 16h + e); math runs on
// v_wmma_f32_16x16x32_f16 with f32 accumulation.
// ---------------------------------------------------------------------------

typedef __attribute__((ext_vector_type(16))) _Float16 v16h;
typedef __attribute__((ext_vector_type(8)))  float    v8f;
typedef __attribute__((ext_vector_type(2)))  __fp16   v2hf;   // cvt_pkrtz result type

#define UNITS   14336
#define INFEAT  4096
#define PACKEDW 1024          // int32 elements (1 crumb-byte each) per unit
#define CLIPV   100.0f
#define KSTEPS  128           // 4096 / 32
#define KC      32            // K-steps per LDS chunk (64 KB staged x)
#define CHUNKS  (KSTEPS / KC)
#define LDS_HALVES (2 * KC * 32 * 16)   // [mtile][kstep][lane][e]

// Byte LUT: crumb c -> f16 high byte. c=0:0x00(0), 1:0x3C(+1), 2:0xBC(-1), 3:0x00(0)
#define TLUT 0x00BC3C00u
// t*MAGIC puts t[1:0] at bits 9:8 and t[3:2] at bits 25:24 (t is 0..255).
#define MAGIC 0x00400100u

typedef union { uint32_t u[8]; v16h h; } BFrag;

// Decode one crumb-byte t (0..255) -> two packed half2 (4 f16 weights).
__device__ __forceinline__ void dec_byte(uint32_t t, uint32_t& lo, uint32_t& hi) {
  uint32_t m0   = t * MAGIC;
  uint32_t selA = (m0 & 0x03000300u) | 0x00030003u;          // crumbs 0,1
  uint32_t selB = ((m0 >> 4) & 0x03000300u) | 0x00030003u;   // crumbs 2,3
  lo = __builtin_amdgcn_perm(TLUT, TLUT, selA);
  hi = __builtin_amdgcn_perm(TLUT, TLUT, selB);
}

__global__ __launch_bounds__(256, 1) void ternary_wmma_kernel(
    const float* __restrict__ x, const int* __restrict__ pw,
    const float* __restrict__ scale, const float* __restrict__ bias,
    float* __restrict__ out) {
  __shared__ alignas(16) _Float16 ldsx[LDS_HALVES];

  const int tid   = threadIdx.x;
  const int lane  = tid & 31;
  const int wave  = tid >> 5;
  const int utile = blockIdx.x * 8 + wave;   // 0..895
  const int ucol  = utile * 16 + (lane & 15);
  const int wsel  = lane >> 4;               // B frag: 0 -> K0-15, 1 -> K16-31

  v8f acc0 = {};  // batch rows 0..15
  v8f acc1 = {};  // batch rows 16..31

  const _Float16* lA0 = ldsx + lane * 16;
  const _Float16* lA1 = ldsx + (KC * 32 + lane) * 16;

  // 4-deep weight prefetch pipeline: each int4 = 16 K-crumbs for this lane.
  const int* pwbase = pw + ucol * PACKEDW + wsel * 4;
  int4 pA = *(const int4*)(pwbase + 0 * 8);
  int4 pB = *(const int4*)(pwbase + 1 * 8);
  int4 pC = *(const int4*)(pwbase + 2 * 8);
  int4 pD = *(const int4*)(pwbase + 3 * 8);

  BFrag bfA, bfB;  // alternating B buffers (avoid WMMA->VALU WAR hazard NOPs)

#define STEP(KS, P, BF)                                                        \
  {                                                                            \
    v16h a0 = *(const v16h*)(lA0 + ((KS) & (KC - 1)) * (32 * 16));             \
    v16h a1 = *(const v16h*)(lA1 + ((KS) & (KC - 1)) * (32 * 16));             \
    dec_byte((uint32_t)P.x, BF.u[0], BF.u[1]);                                 \
    dec_byte((uint32_t)P.y, BF.u[2], BF.u[3]);                                 \
    dec_byte((uint32_t)P.z, BF.u[4], BF.u[5]);                                 \
    dec_byte((uint32_t)P.w, BF.u[6], BF.u[7]);                                 \
    int nks = ((KS) + 4 < KSTEPS) ? (KS) + 4 : KSTEPS - 1;                     \
    P = *(const int4*)(pwbase + nks * 8);                                      \
    acc0 = __builtin_amdgcn_wmma_f32_16x16x32_f16(false, a0, false, BF.h,      \
                                                  (short)0, acc0, false, false); \
    acc1 = __builtin_amdgcn_wmma_f32_16x16x32_f16(false, a1, false, BF.h,      \
                                                  (short)0, acc1, false, false); \
  }

  for (int chunk = 0; chunk < CHUNKS; ++chunk) {
    __syncthreads();
    // Stage x chunk f32->f16 into LDS in a contiguous-K A fragment order, matching B
    // (lanes 0-15: K0-15; lanes 16-31: K16-31), 8 floats/item:
    // two float4 loads -> 4x v_cvt_pkrtz -> one ds_store_b128.
    for (int w = tid; w < 2 * KC * 32 * 2; w += 256) {
      int eg  = w & 1;            // e 0-7 vs 8-15
      int fl  = (w >> 1) & 31;    // fragment lane
      int ksl = (w >> 6) & (KC - 1);
      int mt  = w >> 11;
      int m   = mt * 16 + (fl & 15);
      int kb  = (chunk * KC + ksl) * 32 + eg * 8 + ((fl >> 4) << 4);
      const float4* xp = (const float4*)(x + m * INFEAT + kb);
      float4 f0 = xp[0];
      float4 f1 = xp[1];
      union { v2hf h[4]; int4 v; } st;
      st.h[0] = __builtin_amdgcn_cvt_pkrtz(f0.x, f0.y);
      st.h[1] = __builtin_amdgcn_cvt_pkrtz(f0.z, f0.w);
      st.h[2] = __builtin_amdgcn_cvt_pkrtz(f1.x, f1.y);
      st.h[3] = __builtin_amdgcn_cvt_pkrtz(f1.z, f1.w);
      *(int4*)(ldsx + ((mt * KC + ksl) * 32 + fl) * 16 + eg * 8) = st.v;
    }
    __syncthreads();

    const int ksbase = chunk * KC;
    #pragma unroll 2
    for (int ksl = 0; ksl < KC; ksl += 4) {
      STEP(ksbase + ksl + 0, pA, bfA)
      STEP(ksbase + ksl + 1, pB, bfB)
      STEP(ksbase + ksl + 2, pC, bfA)
      STEP(ksbase + ksl + 3, pD, bfB)
    }
  }
#undef STEP

  // Epilogue: C/D layout is lane 0-15 -> (M=r, N=lane), lane 16-31 -> (M=8+r).
  const float sc   = scale[ucol];
  const float bi   = bias[ucol];
  const int   mrow = (lane >> 4) * 8;
  #pragma unroll
  for (int r = 0; r < 8; ++r) {
    float v0 = fminf(fmaxf(acc0[r] * sc + bi, -CLIPV), CLIPV);
    float v1 = fminf(fmaxf(acc1[r] * sc + bi, -CLIPV), CLIPV);
    out[(mrow + r) * UNITS + ucol]      = v0;
    out[(16 + mrow + r) * UNITS + ucol] = v1;
  }
}

extern "C" void kernel_launch(void* const* d_in, const int* in_sizes, int n_in,
                              void* d_out, int out_size, void* d_ws, size_t ws_size,
                              hipStream_t stream) {
  (void)in_sizes; (void)n_in; (void)d_ws; (void)ws_size; (void)out_size;
  const float* x     = (const float*)d_in[0];
  const int*   pw    = (const int*)d_in[1];
  const float* scale = (const float*)d_in[2];
  const float* bias  = (const float*)d_in[3];
  float*       out   = (float*)d_out;

  // 896 N-tiles / 8 waves per block -> 112 blocks of 256 threads (8 wave32s).
  ternary_wmma_kernel<<<dim3(112), dim3(256), 0, stream>>>(x, pw, scale, bias, out);
}
